// HeteroEdgeAwareTransformer_24017457119609
// MI455X (gfx1250) — matrix operand order repaired and checked by reading, not compiled, not run on hardware
//
#include <hip/hip_runtime.h>
#include <hip/hip_bf16.h>
#include <cstdint>

// ---------------- problem constants (from reference) ----------------
#define NG_  40000
#define NR_  10000
#define H_   128
#define HD_  512
#define L_   3
#define EGG_ 120000
#define ERG_ 40000
#define EGR_ 40000

typedef __attribute__((ext_vector_type(16))) _Float16 v16h;
typedef __attribute__((ext_vector_type(4)))  _Float16 v4h;
typedef __attribute__((ext_vector_type(8)))  float    v8f;
typedef __attribute__((ext_vector_type(4)))  float    v4f;

__device__ __forceinline__ float warp_sum(float v) {
#pragma unroll
  for (int off = 16; off; off >>= 1) v += __shfl_xor(v, off, 32);
  return v;
}

// ============================================================================
// WMMA GEMM:  C[M,N] = act( A[M,K] @ W[K,N] + bias[N]  (+ C if ACC) )
// A,C fp32 row-major; W pre-converted f16 row-major. f32 accumulate.
// block = 128 threads (4 waves); wave -> 16x32 (two 16x16 tiles, A reused);
// block tile = 16x128. Requires M%16==0, K%32==0, N%128==0 (all call sites).
// ACT: 0=none, 1=relu, 2=elu
// ============================================================================
template <int ACT, bool ACC>
__global__ void gemm_wmma_f16(const float* __restrict__ A,
                              const _Float16* __restrict__ Wh,
                              const float* __restrict__ bias,
                              float* __restrict__ C,
                              int M, int K, int N) {
  __shared__ _Float16 As[16 * 32];
  const int lane = threadIdx.x & 31;
  const int wv   = threadIdx.x >> 5;
  const int m0   = blockIdx.x * 16;
  const int ncol = blockIdx.y * 128 + wv * 32;

  // each thread stages 4 contiguous A elements: one b128 load, one b64 LDS store
  const int a_row = threadIdx.x >> 3;        // 0..15
  const int a_col = (threadIdx.x & 7) * 4;   // 0,4,...,28

  v8f c0 = {}, c1 = {};
  for (int kb = 0; kb < K; kb += 32) {
    __syncthreads();
    {
      const v4f av = *reinterpret_cast<const v4f*>(
          A + (size_t)(m0 + a_row) * K + kb + a_col);
      v4h hv;
      hv[0] = (_Float16)av[0]; hv[1] = (_Float16)av[1];
      hv[2] = (_Float16)av[2]; hv[3] = (_Float16)av[3];
      *reinterpret_cast<v4h*>(&As[a_row * 32 + a_col]) = hv;
    }
    __syncthreads();

    // A operand per ISA 16-bit 16x32 layout (two contiguous 16B runs per lane)
    v16h a, b0, b1;
    const int mrow = lane & 15;
    const int koff = (lane & 16) >> 1;       // lanes>=16 handle K+8 / K+24 groups
#pragma unroll
    for (int h = 0; h < 16; ++h) {
      int kl = ((h < 8) ? h : h + 8) + koff;
      a[h] = As[mrow * 32 + kl];
    }
    // B operands: lane&15 = N column, elements = K in same order as A; b1 is N+16
    const _Float16* wr = Wh + (size_t)kb * N + ncol + (lane & 15);
#pragma unroll
    for (int h = 0; h < 16; ++h) {
      int kr = ((h < 8) ? h : h + 8) + koff;
      b0[h] = wr[(size_t)kr * N]; b1[h] = wr[(size_t)kr * N + 16];
    }

    c0 = __builtin_amdgcn_wmma_f32_16x16x32_f16(false, a, false, b0,
                                                (short)0, c0, false, false);
    c1 = __builtin_amdgcn_wmma_f32_16x16x32_f16(false, a, false, b1,
                                                (short)0, c1, false, false);
  }

  // epilogue per f32 C/D layout: VGPR r -> M = r + 8*(lane>=16), N = lane&15
  const int mbase = m0 + ((lane & 16) >> 1);
  auto store_tile = [&](const v8f& c, int ncol_t) {
    const int n  = ncol_t + (lane & 15);
    const float bv = bias[n];
#pragma unroll
    for (int r = 0; r < 8; ++r) {
      size_t ofs = (size_t)(mbase + r) * N + n;
      float val = c[r] + bv;
      if (ACC) val += C[ofs];
      if (ACT == 1)      val = fmaxf(val, 0.f);
      else if (ACT == 2) val = (val > 0.f) ? val : (__expf(val) - 1.f);
      C[ofs] = val;
    }
  };
  store_tile(c0, ncol);
  store_tile(c1, ncol + 16);
}

// ---------------- small utility kernels ----------------
__global__ void fill_f32(float* p, float v, int n) {
  int i = blockIdx.x * blockDim.x + threadIdx.x;
  if (i < n) p[i] = v;
}
__global__ void fill_u32(unsigned* p, unsigned v, int n) {
  int i = blockIdx.x * blockDim.x + threadIdx.x;
  if (i < n) p[i] = v;
}
__global__ void cvt_f32_f16(const float* __restrict__ in, _Float16* __restrict__ out,
                            int n) {
  int i = blockIdx.x * blockDim.x + threadIdx.x;
  if (i < n) out[i] = (_Float16)in[i];
}

// y = relu(x @ w + b), small K (16 or 8), N=128 columns
__global__ void in_proj(const float* __restrict__ X, const float* __restrict__ Wt,
                        const float* __restrict__ B, float* __restrict__ Y,
                        int Nn, int Kin) {
  int idx = blockIdx.x * blockDim.x + threadIdx.x;
  if (idx >= Nn * H_) return;
  int node = idx >> 7, col = idx & 127;
  const float* xr = X + (size_t)node * Kin;
  float acc = B[col];
  for (int i = 0; i < Kin; ++i) acc += xr[i] * Wt[i * H_ + col];
  Y[idx] = fmaxf(acc, 0.f);
}

// ea3[e, 0:3] = ea[e,:] @ epw + epb   (Kin = 3 or 2)
__global__ void ea_proj3(const float* __restrict__ EA, const float* __restrict__ W,
                         const float* __restrict__ B, float* __restrict__ OUT,
                         int E, int Kin) {
  int e = blockIdx.x * blockDim.x + threadIdx.x;
  if (e >= E) return;
  for (int j = 0; j < 3; ++j) {
    float acc = B[j];
    for (int i = 0; i < Kin; ++i) acc += EA[(size_t)e * Kin + i] * W[i * 3 + j];
    OUT[(size_t)e * 3 + j] = acc;
  }
}

__device__ __forceinline__ unsigned f32_key(float x) {
  unsigned u = __float_as_uint(x);
  return (u & 0x80000000u) ? ~u : (u | 0x80000000u);
}
__device__ __forceinline__ float f32_unkey(unsigned k) {
  return (k & 0x80000000u) ? __uint_as_float(k ^ 0x80000000u)
                           : __uint_as_float(~k);
}

// one wave per edge: score[h] = q[dst]·(k[src]+e) / sqrt(128); atomicMax per dst
__global__ void edge_score(const float* __restrict__ Q, const float* __restrict__ Kf,
                           const float* __restrict__ we, const int* __restrict__ ei,
                           const float* __restrict__ ea3, unsigned* __restrict__ mkey,
                           float* __restrict__ score, int E) {
  int wid  = (blockIdx.x * blockDim.x + threadIdx.x) >> 5;
  int lane = threadIdx.x & 31;
  if (wid >= E) return;
  int src = ei[wid], dst = ei[E + wid];
  float e0 = ea3[wid * 3], e1 = ea3[wid * 3 + 1], e2 = ea3[wid * 3 + 2];
  const float* q = Q  + (size_t)dst * HD_;
  const float* k = Kf + (size_t)src * HD_;
#pragma unroll
  for (int h = 0; h < 4; ++h) {
    float p = 0.f;
#pragma unroll
    for (int t = 0; t < 4; ++t) {
      int d = h * 128 + t * 32 + lane;
      float ev = e0 * we[d] + e1 * we[HD_ + d] + e2 * we[2 * HD_ + d];
      p += q[d] * (k[d] + ev);
    }
    p = warp_sum(p);
    if (lane == 0) {
      float s = p * 0.08838834764831843f;   // 1/sqrt(128)
      score[wid * 4 + h] = s;
      atomicMax(&mkey[dst * 4 + h], f32_key(s));
    }
  }
}

// one thread per (edge, head): ex = exp(score - max); atomicAdd denominator
__global__ void edge_exp(const int* __restrict__ ei, const unsigned* __restrict__ mkey,
                         float* __restrict__ score_ex, float* __restrict__ msum, int E) {
  int idx = blockIdx.x * blockDim.x + threadIdx.x;
  if (idx >= E * 4) return;
  int e = idx >> 2, h = idx & 3;
  int dst = ei[E + e];
  float m = f32_unkey(mkey[dst * 4 + h]);
  float v = __expf(score_ex[idx] - m);
  score_ex[idx] = v;
  atomicAdd(&msum[dst * 4 + h], v);
}

// one wave per edge: agg[dst] += alpha[h] * (v[src] + e)
__global__ void edge_msg(const float* __restrict__ V, const float* __restrict__ we,
                         const int* __restrict__ ei, const float* __restrict__ ea3,
                         const float* __restrict__ ex, const float* __restrict__ msum,
                         float* __restrict__ agg, int E) {
  int wid  = (blockIdx.x * blockDim.x + threadIdx.x) >> 5;
  int lane = threadIdx.x & 31;
  if (wid >= E) return;
  int src = ei[wid], dst = ei[E + wid];
  float e0 = ea3[wid * 3], e1 = ea3[wid * 3 + 1], e2 = ea3[wid * 3 + 2];
  float al[4];
#pragma unroll
  for (int h = 0; h < 4; ++h)
    al[h] = ex[wid * 4 + h] / (msum[dst * 4 + h] + 1e-16f);
  const float* vr = V + (size_t)src * HD_;
  float* ar = agg + (size_t)dst * HD_;
#pragma unroll
  for (int j = 0; j < 16; ++j) {
    int d = j * 32 + lane;
    int h = j >> 2;                         // constant after unroll
    float ev = e0 * we[d] + e1 * we[HD_ + d] + e2 * we[2 * HD_ + d];
    atomicAdd(&ar[d], al[h] * (vr[d] + ev));
  }
}

// one wave per node: x = layernorm(x + h) * g + b  (dim = 128)
__global__ void resid_ln(float* __restrict__ x, const float* __restrict__ hadd,
                         const float* __restrict__ g, const float* __restrict__ b,
                         int Nn) {
  int node = (blockIdx.x * blockDim.x + threadIdx.x) >> 5;
  int lane = threadIdx.x & 31;
  if (node >= Nn) return;
  const float* xr = x + (size_t)node * H_;
  const float* hr = hadd + (size_t)node * H_;
  float v[4], s = 0.f;
#pragma unroll
  for (int t = 0; t < 4; ++t) { v[t] = xr[t * 32 + lane] + hr[t * 32 + lane]; s += v[t]; }
  float mu = warp_sum(s) * (1.f / 128.f);
  float q = 0.f;
#pragma unroll
  for (int t = 0; t < 4; ++t) { float d = v[t] - mu; q += d * d; }
  float var = warp_sum(q) * (1.f / 128.f);
  float r = rsqrtf(var + 1e-5f);
#pragma unroll
  for (int t = 0; t < 4; ++t) {
    int d = t * 32 + lane;
    x[(size_t)node * H_ + d] = (v[t] - mu) * r * g[d] + b[d];
  }
}

// one wave per node: out[node] = h_relu[node,:] @ w(128x1) + b
__global__ void head_out(const float* __restrict__ hrelu, const float* __restrict__ w,
                         const float* __restrict__ b, float* __restrict__ out, int Nn) {
  int node = (blockIdx.x * blockDim.x + threadIdx.x) >> 5;
  int lane = threadIdx.x & 31;
  if (node >= Nn) return;
  float p = 0.f;
#pragma unroll
  for (int t = 0; t < 4; ++t) {
    int d = t * 32 + lane;
    p += hrelu[(size_t)node * H_ + d] * w[d];
  }
  p = warp_sum(p);
  if (lane == 0) out[node] = p + b[0];
}

// ---------------- host-side helpers ----------------
static inline void gemm(hipStream_t s, const float* A, const _Float16* Wh,
                        const float* B, float* C, int M, int K, int N,
                        int act, int acc) {
  dim3 g(M / 16, N / 128), blk(128);
  if (acc)            gemm_wmma_f16<0, true ><<<g, blk, 0, s>>>(A, Wh, B, C, M, K, N);
  else if (act == 0)  gemm_wmma_f16<0, false><<<g, blk, 0, s>>>(A, Wh, B, C, M, K, N);
  else if (act == 1)  gemm_wmma_f16<1, false><<<g, blk, 0, s>>>(A, Wh, B, C, M, K, N);
  else                gemm_wmma_f16<2, false><<<g, blk, 0, s>>>(A, Wh, B, C, M, K, N);
}
static inline void cvt(hipStream_t s, const float* in, _Float16* out, int n) {
  cvt_f32_f16<<<(n + 255) / 256, 256, 0, s>>>(in, out, n);
}

extern "C" void kernel_launch(void* const* d_in, const int* in_sizes, int n_in,
                              void* d_out, int out_size, void* d_ws, size_t ws_size,
                              hipStream_t stream) {
  (void)in_sizes; (void)n_in; (void)out_size; (void)ws_size;
  const float* x_grid  = (const float*)d_in[0];
  const float* x_river = (const float*)d_in[1];
  const int*   ei_gg   = (const int*)d_in[2];
  const int*   ei_rg   = (const int*)d_in[3];
  const int*   ei_gr   = (const int*)d_in[4];
  const float* ea_gg   = (const float*)d_in[5];
  const float* ea_rg   = (const float*)d_in[6];
  const float* ea_gr   = (const float*)d_in[7];
  int p = 8;  // params dict, insertion order
  const float* inw_g  = (const float*)d_in[p++];
  const float* inb_g  = (const float*)d_in[p++];
  const float* inw_r  = (const float*)d_in[p++];
  const float* inb_r  = (const float*)d_in[p++];
  const float* epw_gg = (const float*)d_in[p++];
  const float* epb_gg = (const float*)d_in[p++];
  const float* epw_rg = (const float*)d_in[p++];
  const float* epb_rg = (const float*)d_in[p++];
  const float* epw_gr = (const float*)d_in[p++];
  const float* epb_gr = (const float*)d_in[p++];
  const float* wq     = (const float*)d_in[p++];
  const float* bq     = (const float*)d_in[p++];
  const float* wk     = (const float*)d_in[p++];
  const float* bk     = (const float*)d_in[p++];
  const float* wv     = (const float*)d_in[p++];
  const float* bv     = (const float*)d_in[p++];
  const float* weA    = (const float*)d_in[p++];
  const float* wskip  = (const float*)d_in[p++];
  const float* bskip  = (const float*)d_in[p++];
  const float* proj_w = (const float*)d_in[p++];
  const float* proj_b = (const float*)d_in[p++];
  const float* ln_g   = (const float*)d_in[p++];
  const float* ln_b   = (const float*)d_in[p++];
  const float* h1w    = (const float*)d_in[p++];
  const float* h1b    = (const float*)d_in[p++];
  const float* h2w    = (const float*)d_in[p++];
  const float* h2b    = (const float*)d_in[p++];

  // -------- workspace carve (~415 MB) --------
  float* fw = (float*)d_ws;
  size_t o = 0;
  float* xg   = fw + o; o += (size_t)NG_ * H_;
  float* xr   = fw + o; o += (size_t)NR_ * H_;
  float* Qb   = fw + o; o += (size_t)NG_ * HD_;
  float* Kbuf = fw + o; o += (size_t)NG_ * HD_;
  float* Vbuf = fw + o; o += (size_t)NG_ * HD_;
  float* aggG = fw + o; o += (size_t)NG_ * HD_;
  float* aggR = fw + o; o += (size_t)NR_ * HD_;
  float* hG   = fw + o; o += (size_t)NG_ * H_;
  float* hR   = fw + o; o += (size_t)NR_ * H_;
  float* msum = fw + o; o += (size_t)NG_ * 4;
  unsigned* mkey = (unsigned*)(fw + o); o += (size_t)NG_ * 4;
  float* exb  = fw + o; o += (size_t)EGG_ * 4;
  float* ea3g = fw + o; o += (size_t)EGG_ * 3;
  float* ea3r = fw + o; o += (size_t)ERG_ * 3;
  float* ea3R = fw + o; o += (size_t)EGR_ * 3;
  // f16 weight mirrors
  const int WBIG = L_ * 3 * H_ * HD_;     // 589824
  const int WPJ  = L_ * 2 * HD_ * H_;     // 393216
  const int WH1  = 2 * H_ * H_;           // 32768
  _Float16* h16 = (_Float16*)(fw + o);
  size_t ho = 0;
  _Float16* wq16 = h16 + ho; ho += WBIG;
  _Float16* wk16 = h16 + ho; ho += WBIG;
  _Float16* wv16 = h16 + ho; ho += WBIG;
  _Float16* ws16 = h16 + ho; ho += WBIG;
  _Float16* pj16 = h16 + ho; ho += WPJ;
  _Float16* h116 = h16 + ho; ho += WH1;

  const unsigned NEG_INF_KEY = 0x007FFFFFu;   // key(-inf)

  // -------- one-time weight conversion fp32 -> f16 --------
  cvt(stream, wq, wq16, WBIG);
  cvt(stream, wk, wk16, WBIG);
  cvt(stream, wv, wv16, WBIG);
  cvt(stream, wskip, ws16, WBIG);
  cvt(stream, proj_w, pj16, WPJ);
  cvt(stream, h1w, h116, WH1);

  // -------- edge-attr projections (layer-independent) --------
  ea_proj3<<<(EGG_ + 255) / 256, 256, 0, stream>>>(ea_gg, epw_gg, epb_gg, ea3g, EGG_, 3);
  ea_proj3<<<(ERG_ + 255) / 256, 256, 0, stream>>>(ea_rg, epw_rg, epb_rg, ea3r, ERG_, 2);
  ea_proj3<<<(EGR_ + 255) / 256, 256, 0, stream>>>(ea_gr, epw_gr, epb_gr, ea3R, EGR_, 2);

  // -------- input projections + relu --------
  in_proj<<<(NG_ * H_ + 255) / 256, 256, 0, stream>>>(x_grid,  inw_g, inb_g, xg, NG_, 16);
  in_proj<<<(NR_ * H_ + 255) / 256, 256, 0, stream>>>(x_river, inw_r, inb_r, xr, NR_, 8);

  auto off_w  = [](int l, int et) { return (size_t)(l * 3 + et) * H_ * HD_; };
  auto off_b  = [](int l, int et) { return (size_t)(l * 3 + et) * HD_; };
  auto off_we = [](int l, int et) { return (size_t)(l * 3 + et) * 3 * HD_; };

  for (int l = 0; l < L_; ++l) {
    // ===== grid-destination: aggG initialized with both skip terms =====
    gemm(stream, xg, ws16 + off_w(l, 0), bskip + off_b(l, 0), aggG, NG_, H_, HD_, 0, 0);
    gemm(stream, xg, ws16 + off_w(l, 1), bskip + off_b(l, 1), aggG, NG_, H_, HD_, 0, 1);

    // ---- conv gg (grid -> grid) ----
    gemm(stream, xg, wq16 + off_w(l, 0), bq + off_b(l, 0), Qb,   NG_, H_, HD_, 0, 0);
    gemm(stream, xg, wk16 + off_w(l, 0), bk + off_b(l, 0), Kbuf, NG_, H_, HD_, 0, 0);
    gemm(stream, xg, wv16 + off_w(l, 0), bv + off_b(l, 0), Vbuf, NG_, H_, HD_, 0, 0);
    fill_u32<<<(NG_ * 4 + 255) / 256, 256, 0, stream>>>(mkey, NEG_INF_KEY, NG_ * 4);
    fill_f32<<<(NG_ * 4 + 255) / 256, 256, 0, stream>>>(msum, 0.f, NG_ * 4);
    edge_score<<<(EGG_ + 7) / 8, 256, 0, stream>>>(Qb, Kbuf, weA + off_we(l, 0),
                                                   ei_gg, ea3g, mkey, exb, EGG_);
    edge_exp<<<(EGG_ * 4 + 255) / 256, 256, 0, stream>>>(ei_gg, mkey, exb, msum, EGG_);
    edge_msg<<<(EGG_ + 7) / 8, 256, 0, stream>>>(Vbuf, weA + off_we(l, 0),
                                                 ei_gg, ea3g, exb, msum, aggG, EGG_);

    // ---- conv rg (river -> grid) ----
    gemm(stream, xg, wq16 + off_w(l, 1), bq + off_b(l, 1), Qb,   NG_, H_, HD_, 0, 0);
    gemm(stream, xr, wk16 + off_w(l, 1), bk + off_b(l, 1), Kbuf, NR_, H_, HD_, 0, 0);
    gemm(stream, xr, wv16 + off_w(l, 1), bv + off_b(l, 1), Vbuf, NR_, H_, HD_, 0, 0);
    fill_u32<<<(NG_ * 4 + 255) / 256, 256, 0, stream>>>(mkey, NEG_INF_KEY, NG_ * 4);
    fill_f32<<<(NG_ * 4 + 255) / 256, 256, 0, stream>>>(msum, 0.f, NG_ * 4);
    edge_score<<<(ERG_ + 7) / 8, 256, 0, stream>>>(Qb, Kbuf, weA + off_we(l, 1),
                                                   ei_rg, ea3r, mkey, exb, ERG_);
    edge_exp<<<(ERG_ * 4 + 255) / 256, 256, 0, stream>>>(ei_rg, mkey, exb, msum, ERG_);
    edge_msg<<<(ERG_ + 7) / 8, 256, 0, stream>>>(Vbuf, weA + off_we(l, 1),
                                                 ei_rg, ea3r, exb, msum, aggG, ERG_);

    // ---- conv gr (grid -> river) ----
    gemm(stream, xr, ws16 + off_w(l, 2), bskip + off_b(l, 2), aggR, NR_, H_, HD_, 0, 0);
    gemm(stream, xr, wq16 + off_w(l, 2), bq + off_b(l, 2), Qb,   NR_, H_, HD_, 0, 0);
    gemm(stream, xg, wk16 + off_w(l, 2), bk + off_b(l, 2), Kbuf, NG_, H_, HD_, 0, 0);
    gemm(stream, xg, wv16 + off_w(l, 2), bv + off_b(l, 2), Vbuf, NG_, H_, HD_, 0, 0);
    fill_u32<<<(NR_ * 4 + 255) / 256, 256, 0, stream>>>(mkey, NEG_INF_KEY, NR_ * 4);
    fill_f32<<<(NR_ * 4 + 255) / 256, 256, 0, stream>>>(msum, 0.f, NR_ * 4);
    edge_score<<<(EGR_ + 7) / 8, 256, 0, stream>>>(Qb, Kbuf, weA + off_we(l, 2),
                                                   ei_gr, ea3R, mkey, exb, EGR_);
    edge_exp<<<(EGR_ * 4 + 255) / 256, 256, 0, stream>>>(ei_gr, mkey, exb, msum, EGR_);
    edge_msg<<<(EGR_ + 7) / 8, 256, 0, stream>>>(Vbuf, weA + off_we(l, 2),
                                                 ei_gr, ea3R, exb, msum, aggR, EGR_);

    // ---- proj -> elu -> residual layernorm ----
    gemm(stream, aggG, pj16 + (size_t)(l * 2 + 0) * HD_ * H_,
         proj_b + (size_t)(l * 2 + 0) * H_, hG, NG_, HD_, H_, 2, 0);
    gemm(stream, aggR, pj16 + (size_t)(l * 2 + 1) * HD_ * H_,
         proj_b + (size_t)(l * 2 + 1) * H_, hR, NR_, HD_, H_, 2, 0);
    resid_ln<<<(NG_ + 7) / 8, 256, 0, stream>>>(xg, hG, ln_g + (size_t)(l * 2 + 0) * H_,
                                                ln_b + (size_t)(l * 2 + 0) * H_, NG_);
    resid_ln<<<(NR_ + 7) / 8, 256, 0, stream>>>(xr, hR, ln_g + (size_t)(l * 2 + 1) * H_,
                                                ln_b + (size_t)(l * 2 + 1) * H_, NR_);
  }

  // -------- output heads --------
  float* out = (float*)d_out;
  gemm(stream, xg, h116, h1b, hG, NG_, H_, H_, 1, 0);
  head_out<<<(NG_ + 7) / 8, 256, 0, stream>>>(hG, h2w, h2b, out, NG_);
  gemm(stream, xr, h116 + (size_t)H_ * H_, h1b + H_, hR, NR_, H_, H_, 1, 0);
  head_out<<<(NR_ + 7) / 8, 256, 0, stream>>>(hR, h2w + H_, h2b + 1, out + NG_, NR_);
}
